// MultiHeadAttention_4587025072515
// MI455X (gfx1250) — matrix operand order repaired and checked by reading, not compiled, not run on hardware
//
#include <hip/hip_runtime.h>
#include <hip/hip_bf16.h>

#define HIDDEN 2048
#define NHEADS 16
#define HDIM   128
#define BATCH  2
#define SEQ    2048
#define MTOT   (BATCH * SEQ)   // 4096

typedef __attribute__((ext_vector_type(16))) __bf16 bf16x16;
typedef __attribute__((ext_vector_type(8)))  float  f32x8;

union BFrag { bf16x16 f; unsigned int u[8]; };

__device__ __forceinline__ unsigned short f2bf(float x) {
    unsigned int u = __float_as_uint(x);
    u += 0x7FFFu + ((u >> 16) & 1u);   // round-to-nearest-even
    return (unsigned short)(u >> 16);
}

// --- CDNA5 async copy global->LDS (ASYNCcnt-tracked, no VGPR staging) ------
__device__ __forceinline__ void async_b128(unsigned lds_addr, unsigned goff,
                                           const void* sbase) {
    // GVS mode: mem_addr = SGPR64 + VGPR32 offset ; vdst holds LDS byte addr
    asm volatile("global_load_async_to_lds_b128 %0, %1, %2"
                 :: "v"(lds_addr), "v"(goff), "s"(sbase) : "memory");
}
__device__ __forceinline__ void wait_async0() {
    asm volatile("s_wait_asynccnt 0x0" ::: "memory");
}
__device__ __forceinline__ void wait_async4() {
    asm volatile("s_wait_asynccnt 0x4" ::: "memory");
}
__device__ __forceinline__ unsigned lds_addr_of(const void* p) {
    return (unsigned)(size_t)p;   // low 32 bits of flat LDS addr = DS offset
}

// ---------------------------------------------------------------------------
// fp32 -> bf16 conversion (vectorized x4)
// ---------------------------------------------------------------------------
__global__ void cvt_f32_bf16(const float* __restrict__ src,
                             unsigned short* __restrict__ dst, int n4) {
    int i = blockIdx.x * blockDim.x + threadIdx.x;
    if (i < n4) {
        float4 v = ((const float4*)src)[i];
        unsigned int lo = (unsigned int)f2bf(v.x) | ((unsigned int)f2bf(v.y) << 16);
        unsigned int hi = (unsigned int)f2bf(v.z) | ((unsigned int)f2bf(v.w) << 16);
        ((uint2*)dst)[i] = make_uint2(lo, hi);
    }
}

// ---------------------------------------------------------------------------
// Tiled WMMA GEMM:  C[M,N] = A[M,K] * W[N,K]^T   (bf16 in, f32 acc)
// Block tile 128x128x32, 8 waves, wave tile 32x64 (2x4 WMMA).
// Double-buffered LDS fed by async-to-LDS copies; s_wait_asynccnt<=4 keeps
// the next tile's 4 copies in flight while the current tile is consumed.
// ---------------------------------------------------------------------------
#define BM 128
#define BN 128
#define BK 32
#define LDA 40   // LDS row stride in ushorts (32 + 8 pad; 80B, 16B-aligned)
#define BUFSH (BM * LDA)          // ushorts per buffer
#define BUFBYTES (BUFSH * 2)      // bytes per buffer

__global__ void __launch_bounds__(256)
gemm_bf16(const unsigned short* __restrict__ A,   // [M,K] bf16
          const unsigned short* __restrict__ W,   // [N,K] bf16
          unsigned short* __restrict__ outB,      // bf16 output (or null)
          float* __restrict__ outF,               // f32 output  (or null)
          int M, int N, int K, int storeF)
{
    __shared__ __align__(16) unsigned short As[2 * BUFSH];
    __shared__ __align__(16) unsigned short Ws[2 * BUFSH];

    const int tid  = threadIdx.x;
    const int lane = tid & 31;
    const int wid  = tid >> 5;
    const int half = lane >> 4;
    const int ln   = lane & 15;

    const int bm = blockIdx.y * BM;
    const int bn = blockIdx.x * BN;
    const int wm = (wid & 3) * 32;
    const int wn = (wid >> 2) * 64;

    // per-thread copy slots: 512 16B chunks per operand, 2 per thread
    const int r0 = tid >> 2;
    const int l0 = (tid & 3) * 8;
    const unsigned ldsA0 = lds_addr_of(As) + (unsigned)(r0 * LDA + l0) * 2u;
    const unsigned ldsA1 = ldsA0 + 64u * LDA * 2u;
    const unsigned ldsW0 = lds_addr_of(Ws) + (unsigned)(r0 * LDA + l0) * 2u;
    const unsigned ldsW1 = ldsW0 + 64u * LDA * 2u;
    const unsigned gA0 = (unsigned)(((size_t)(bm + r0) * K + l0) * 2);
    const unsigned gA1 = (unsigned)(((size_t)(bm + r0 + 64) * K + l0) * 2);
    const unsigned gW0 = (unsigned)(((size_t)(bn + r0) * K + l0) * 2);
    const unsigned gW1 = (unsigned)(((size_t)(bn + r0 + 64) * K + l0) * 2);

#define ISSUE_TILE(bi, kelem) do {                                   \
        unsigned bo = (unsigned)(bi) * BUFBYTES;                     \
        unsigned ko = (unsigned)(kelem) * 2u;                        \
        async_b128(ldsA0 + bo, gA0 + ko, A);                         \
        async_b128(ldsA1 + bo, gA1 + ko, A);                         \
        async_b128(ldsW0 + bo, gW0 + ko, W);                         \
        async_b128(ldsW1 + bo, gW1 + ko, W);                         \
    } while (0)

    f32x8 zero = {0.f,0.f,0.f,0.f,0.f,0.f,0.f,0.f};
    f32x8 acc[2][4];
#pragma unroll
    for (int i = 0; i < 2; ++i)
#pragma unroll
        for (int j = 0; j < 4; ++j) acc[i][j] = zero;

    const int nk = K / BK;
    ISSUE_TILE(0, 0);

    for (int kt = 0; kt < nk; ++kt) {
        const int cur = kt & 1;
        if (kt + 1 < nk) {
            ISSUE_TILE(cur ^ 1, (kt + 1) * BK);
            wait_async4();          // current tile's 4 copies done; next in flight
        } else {
            wait_async0();
        }
        __syncthreads();            // tile visible to all waves

        const unsigned short* Ab = &As[cur * BUFSH];
        const unsigned short* Wb = &Ws[cur * BUFSH];

        BFrag a[2], b[4];
#pragma unroll
        for (int mf = 0; mf < 2; ++mf) {
            const unsigned short* base = &Ab[(wm + mf * 16 + ln) * LDA];
#pragma unroll
            for (int v = 0; v < 8; ++v) {
                int k0 = half * 16 + v * 2;
                a[mf].u[v] = *(const unsigned int*)(&base[k0]);
            }
        }
#pragma unroll
        for (int nf = 0; nf < 4; ++nf) {
            const unsigned short* base = &Wb[(wn + nf * 16 + ln) * LDA];
#pragma unroll
            for (int v = 0; v < 8; ++v) {
                int k0 = half * 16 + v * 2;
                b[nf].u[v] = *(const unsigned int*)(&base[k0]);
            }
        }
#pragma unroll
        for (int mf = 0; mf < 2; ++mf)
#pragma unroll
            for (int nf = 0; nf < 4; ++nf)
                acc[mf][nf] = __builtin_amdgcn_wmma_f32_16x16x32_bf16(
                    false, a[mf].f, false, b[nf].f, (short)0, acc[mf][nf],
                    false, false);
        __syncthreads();            // everyone done reading buf before reissue
    }
#undef ISSUE_TILE

#pragma unroll
    for (int mf = 0; mf < 2; ++mf)
#pragma unroll
        for (int nf = 0; nf < 4; ++nf)
#pragma unroll
            for (int i = 0; i < 8; ++i) {
                int gr = bm + wm + mf * 16 + i + half * 8;
                int gc = bn + wn + nf * 16 + ln;
                float val = acc[mf][nf][i];
                if (storeF) outF[(size_t)gr * N + gc] = val;
                else        outB[(size_t)gr * N + gc] = f2bf(val);
            }
}

// ---------------------------------------------------------------------------
// Flash attention: block = (qtile 128, head, batch); 8 waves x 16 q-rows each.
// Q tile and K tiles arrive via async-to-LDS; V is transposed through VGPRs.
// ---------------------------------------------------------------------------
#define QTILE 128
#define KTILE 32
#define LQ 136   // LDS stride for 128-wide rows (+8 pad)
#define LV 40    // LDS stride for 32-wide rows  (+8 pad)

__global__ void __launch_bounds__(256)
flash_attn(const unsigned short* __restrict__ Q,
           const unsigned short* __restrict__ Kd,
           const unsigned short* __restrict__ Vd,
           unsigned short* __restrict__ O)
{
    __shared__ __align__(16) unsigned short Qs[QTILE * LQ];
    __shared__ __align__(16) unsigned short Ks[KTILE * LQ];
    __shared__ __align__(16) unsigned short Vts[HDIM * LV];   // V transposed
    __shared__ __align__(16) unsigned short Ps[8 * 16 * LV];  // per-wave P

    const int tid  = threadIdx.x;
    const int lane = tid & 31;
    const int wid  = tid >> 5;
    const int half = lane >> 4;
    const int ln   = lane & 15;

    const int qbase = blockIdx.x * QTILE;
    const int h     = blockIdx.y;
    const int b     = blockIdx.z;

    const size_t rowstride = HIDDEN;
    const size_t headoff   = (size_t)h * HDIM;
    const size_t baseBS    = (size_t)b * SEQ;

    const unsigned qsb = lds_addr_of(Qs);
    const unsigned ksb = lds_addr_of(Ks);

    // Q tile 128x128 bf16: 2048 16B chunks, 8 async copies per thread
#pragma unroll
    for (int t = 0; t < 8; ++t) {
        int c   = tid + t * 256;
        int row = c >> 4;
        int col = (c & 15) * 8;
        unsigned goff =
            (unsigned)(((baseBS + qbase + row) * rowstride + headoff + col) * 2);
        async_b128(qsb + (unsigned)(row * LQ + col) * 2u, goff, Q);
    }

    f32x8 zero = {0.f,0.f,0.f,0.f,0.f,0.f,0.f,0.f};
    float m_i[8], l_i[8];
    f32x8 o[8];
#pragma unroll
    for (int i = 0; i < 8; ++i) { m_i[i] = -1e30f; l_i[i] = 0.f; o[i] = zero; }

    const float scale = 0.08838834764831845f;   // 1/sqrt(128)
    const int nkt = (qbase + QTILE) / KTILE;    // causal: tiles <= q-tile end

    // per-thread K/V chunk coords (512 chunks, 2 per thread)
    const int kr0 = tid >> 4;            // rows 0..15
    const int kc0 = (tid & 15) * 8;

    for (int kt = 0; kt < nkt; ++kt) {
        const int kb = kt * KTILE;
        __syncthreads();   // previous iteration done with Ks/Vts/Ps

        // K tile 32x128 via async copies (2 per thread)
#pragma unroll
        for (int t = 0; t < 2; ++t) {
            int row = kr0 + t * 16;
            unsigned goff =
                (unsigned)(((baseBS + kb + row) * rowstride + headoff + kc0) * 2);
            async_b128(ksb + (unsigned)(row * LQ + kc0) * 2u, goff, Kd);
        }
        // V tile transposed into Vts[128][32] (VGPR path: needs transpose)
#pragma unroll
        for (int t = 0; t < 2; ++t) {
            int row = kr0 + t * 16;
            uint4 vv = *(const uint4*)(
                &Vd[(baseBS + kb + row) * rowstride + headoff + kc0]);
            const unsigned short* vs = (const unsigned short*)&vv;
#pragma unroll
            for (int j = 0; j < 8; ++j)
                Vts[(kc0 + j) * LV + row] = vs[j];
        }
        wait_async0();     // Q (first iter) + K copies landed in LDS
        __syncthreads();   // + dscnt-waited V stores visible to all waves

        // S = Q * K^T : wave's 16 rows x 32 keys, K-dim = 128 = 4 WMMA steps
        f32x8 s[2] = {zero, zero};
#pragma unroll
        for (int kk = 0; kk < 4; ++kk) {
            BFrag aq;
            const unsigned short* abase = &Qs[(wid * 16 + ln) * LQ + kk * 32];
#pragma unroll
            for (int v = 0; v < 8; ++v) {
                int k0 = half * 16 + v * 2;
                aq.u[v] = *(const unsigned int*)(&abase[k0]);
            }
#pragma unroll
            for (int j = 0; j < 2; ++j) {
                BFrag bk;
                const unsigned short* bbase = &Ks[(j * 16 + ln) * LQ + kk * 32];
#pragma unroll
                for (int v = 0; v < 8; ++v) {
                    int k0 = half * 16 + v * 2;
                    bk.u[v] = *(const unsigned int*)(&bbase[k0]);
                }
                s[j] = __builtin_amdgcn_wmma_f32_16x16x32_bf16(
                    false, aq.f, false, bk.f, (short)0, s[j], false, false);
            }
        }

        // online softmax over this 16x32 score block (per-lane: 8 rows)
#pragma unroll
        for (int i = 0; i < 8; ++i) {
            int qg = qbase + wid * 16 + i + half * 8;
            bool msk0 = (kb + ln)      > qg;
            bool msk1 = (kb + 16 + ln) > qg;
            float v0 = msk0 ? -1e30f : s[0][i] * scale;
            float v1 = msk1 ? -1e30f : s[1][i] * scale;
            float mx = fmaxf(v0, v1);
#pragma unroll
            for (int d = 8; d >= 1; d >>= 1) mx = fmaxf(mx, __shfl_xor(mx, d, 32));
            float m_new = fmaxf(m_i[i], mx);
            float corr  = __expf(m_i[i] - m_new);
            float p0 = msk0 ? 0.f : __expf(v0 - m_new);
            float p1 = msk1 ? 0.f : __expf(v1 - m_new);
            float ps = p0 + p1;
#pragma unroll
            for (int d = 8; d >= 1; d >>= 1) ps += __shfl_xor(ps, d, 32);
            l_i[i] = l_i[i] * corr + ps;
            m_i[i] = m_new;
#pragma unroll
            for (int n = 0; n < 8; ++n) o[n][i] *= corr;
            unsigned short* prow = &Ps[(wid * 16 + i + half * 8) * LV];
            prow[ln]      = f2bf(p0);
            prow[16 + ln] = f2bf(p1);
        }
        __syncthreads();

        // O += P(16x32) * V(32x128) : 8 WMMAs, shared A fragment
        BFrag pf;
        const unsigned short* pbase = &Ps[(wid * 16 + ln) * LV];
#pragma unroll
        for (int v = 0; v < 8; ++v) {
            int k0 = half * 16 + v * 2;
            pf.u[v] = *(const unsigned int*)(&pbase[k0]);
        }
#pragma unroll
        for (int nf = 0; nf < 8; ++nf) {
            BFrag bv;
            const unsigned short* vbase = &Vts[(nf * 16 + ln) * LV];
#pragma unroll
            for (int v = 0; v < 8; ++v) {
                int k0 = half * 16 + v * 2;
                bv.u[v] = *(const unsigned int*)(&vbase[k0]);
            }
            o[nf] = __builtin_amdgcn_wmma_f32_16x16x32_bf16(
                false, pf.f, false, bv.f, (short)0, o[nf], false, false);
        }
    }

    // epilogue: normalize and store bf16
#pragma unroll
    for (int i = 0; i < 8; ++i) {
        float inv = 1.0f / l_i[i];
        int qg = qbase + wid * 16 + i + half * 8;
        size_t rowoff = (baseBS + qg) * rowstride + headoff;
#pragma unroll
        for (int nf = 0; nf < 8; ++nf)
            O[rowoff + nf * 16 + ln] = f2bf(o[nf][i] * inv);
    }
}

// ---------------------------------------------------------------------------
// launch
// ---------------------------------------------------------------------------
extern "C" void kernel_launch(void* const* d_in, const int* in_sizes, int n_in,
                              void* d_out, int out_size, void* d_ws, size_t ws_size,
                              hipStream_t stream)
{
    (void)in_sizes; (void)n_in; (void)out_size; (void)ws_size;
    const float* x  = (const float*)d_in[0];
    const float* wq = (const float*)d_in[1];
    const float* wk = (const float*)d_in[2];
    const float* wv = (const float*)d_in[3];
    const float* wo = (const float*)d_in[4];

    const size_t NX = (size_t)MTOT * HIDDEN;     // 8388608
    const size_t NW = (size_t)HIDDEN * HIDDEN;   // 4194304

    unsigned short* xb  = (unsigned short*)d_ws;
    unsigned short* wqb = xb  + NX;
    unsigned short* wkb = wqb + NW;
    unsigned short* wvb = wkb + NW;
    unsigned short* wob = wvb + NW;
    unsigned short* qb  = wob + NW;
    unsigned short* kb  = qb  + NX;
    unsigned short* vb  = kb  + NX;
    unsigned short* ab  = vb  + NX;

    cvt_f32_bf16<<<(int)(NX / 4 / 256), 256, 0, stream>>>(x,  xb,  (int)(NX / 4));
    cvt_f32_bf16<<<(int)(NW / 4 / 256), 256, 0, stream>>>(wq, wqb, (int)(NW / 4));
    cvt_f32_bf16<<<(int)(NW / 4 / 256), 256, 0, stream>>>(wk, wkb, (int)(NW / 4));
    cvt_f32_bf16<<<(int)(NW / 4 / 256), 256, 0, stream>>>(wv, wvb, (int)(NW / 4));
    cvt_f32_bf16<<<(int)(NW / 4 / 256), 256, 0, stream>>>(wo, wob, (int)(NW / 4));

    dim3 gg(HIDDEN / BN, MTOT / BM);   // (16, 32)
    gemm_bf16<<<gg, 256, 0, stream>>>(xb, wqb, qb, nullptr, MTOT, HIDDEN, HIDDEN, 0);
    gemm_bf16<<<gg, 256, 0, stream>>>(xb, wkb, kb, nullptr, MTOT, HIDDEN, HIDDEN, 0);
    gemm_bf16<<<gg, 256, 0, stream>>>(xb, wvb, vb, nullptr, MTOT, HIDDEN, HIDDEN, 0);

    flash_attn<<<dim3(SEQ / QTILE, NHEADS, BATCH), 256, 0, stream>>>(qb, kb, vb, ab);

    gemm_bf16<<<gg, 256, 0, stream>>>(ab, wob, nullptr, (float*)d_out,
                                      MTOT, HIDDEN, HIDDEN, 1);
}
